// GPT_72138270703788
// MI455X (gfx1250) — matrix operand order repaired and checked by reading, not compiled, not run on hardware
//
#include <hip/hip_runtime.h>
#include <hip/hip_bf16.h>
#include <math.h>

// ---------------------------------------------------------------------------
// Hierarchical GPT forward for MI455X (gfx1250, wave32, WMMA).
//
// Roofline: ~130 GFLOP but ~860MB of f32 weights -> bandwidth bound
// (~37us of HBM traffic at 23.3 TB/s). Strategy: read f32 weights once-ish
// (all weight matrices fit in the 192MB L2), convert to f16 in registers,
// and feed v_wmma_f32_16x16x32_f16. Each wave computes 4 M-tiles x 1 N-tile
// so every weight (B) fragment loaded is reused by 4 WMMAs.
// ---------------------------------------------------------------------------

#define BB      2
#define TT      512
#define DLOC    512
#define DGLB    1024
#define NHEAD   16
#define TGLB    85      // BLOCK // 6
#define NVOCAB  50304
#define NLH     6       // local half layers
#define NLG     12      // global layers
#define LNEPS   1e-5f

typedef __attribute__((ext_vector_type(16))) _Float16 v16h;
typedef __attribute__((ext_vector_type(8)))  float    v8f;

// ------------------------- wave32 reduction --------------------------------
__device__ __forceinline__ float wsum(float v) {
  v += __shfl_xor(v, 16, 32);
  v += __shfl_xor(v, 8, 32);
  v += __shfl_xor(v, 4, 32);
  v += __shfl_xor(v, 2, 32);
  v += __shfl_xor(v, 1, 32);
  return v;
}

// ------------------------- embedding ---------------------------------------
__global__ void embed_kernel(const int* __restrict__ idx,
                             const float* __restrict__ wte,
                             const float* __restrict__ wpe,
                             float* __restrict__ x) {
  int i = blockIdx.x * blockDim.x + threadIdx.x;            // over B*T*DL
  if (i >= BB * TT * DLOC) return;
  int d = i % DLOC;
  int bt = i / DLOC;
  int t = bt % TT;
  x[i] = wte[(size_t)idx[bt] * DLOC + d] + wpe[(size_t)t * DLOC + d];
}

// ------------------------- layernorm (wave per row) ------------------------
__global__ void layernorm_kernel(const float* __restrict__ in,
                                 const float* __restrict__ w,
                                 const float* __restrict__ bvec,
                                 float* __restrict__ out,
                                 int C, long instride) {
  int row = blockIdx.x;
  int lane = threadIdx.x;
  const float* x = in + (long)row * instride;
  float s = 0.f;
  for (int d = lane; d < C; d += 32) s += x[d];
  s = wsum(s);
  float mu = s / (float)C;
  float v = 0.f;
  for (int d = lane; d < C; d += 32) { float t = x[d] - mu; v += t * t; }
  v = wsum(v);
  float inv = rsqrtf(v / (float)C + LNEPS);
  float* o = out + (long)row * C;
  for (int d = lane; d < C; d += 32)
    o[d] = (x[d] - mu) * inv * w[d] + bvec[d];
}

// ------------------------- WMMA GEMM ---------------------------------------
// Y[M,N] = epilogue( X[M,K] @ W[K,N] + bias[N] )   (+ optional GELU, +R[M,N])
// One wave owns 4 vertically stacked 16x16 output tiles in one N-column of
// tiles; B (weight) fragments are loaded once per k-step and reused 4x.
__global__ void __launch_bounds__(256)
gemm_wmma(const float* __restrict__ X, const float* __restrict__ W,
          const float* __restrict__ bias, const float* __restrict__ R,
          float* __restrict__ Y, int M, int N, int K, int dogelu) {
  const int wave = threadIdx.x >> 5;
  const int lane = threadIdx.x & 31;
  const int tilesN = N >> 4;
  const int tn = blockIdx.x * 8 + wave;
  if (tn >= tilesN) return;                 // wave-uniform: EXEC stays all-1s

  const int tm0 = blockIdx.y * 4;           // first of 4 M-tiles for this wave
  const int n0 = tn << 4;
  const int half = lane >> 4;               // 0|1
  const int l16 = lane & 15;

  // A fragment rows (16x32 f16 layout: M = lane&15 in both lane halves)
  const float* Arow[4];
#pragma unroll
  for (int i = 0; i < 4; ++i) {
    int r = (tm0 + i) * 16 + l16;
    if (r >= M) r = M - 1;                  // clamp; extra rows never stored
    Arow[i] = X + (size_t)r * K;
  }
  const int akb = half ? 8 : 0;             // A K sub-offset per lane half
  const int bkoff = half ? 8 : 0;           // B K offset per lane half
  const float* Bcol = W + (n0 + l16);       // B fragment column (N = lane&15)

  v8f c0 = {}, c1 = {}, c2 = {}, c3 = {};

  for (int k0 = 0; k0 < K; k0 += 32) {
    v16h b;
#pragma unroll
    for (int j = 0; j < 16; ++j)
      b[j] = (_Float16)Bcol[(size_t)(k0 + bkoff + (j & 7) + ((j >> 3) << 4)) * N];

    v16h a0, a1, a2, a3;
#pragma unroll
    for (int j = 0; j < 8; ++j) {
      a0[j]     = (_Float16)Arow[0][k0 + akb + j];
      a0[j + 8] = (_Float16)Arow[0][k0 + 16 + akb + j];
      a1[j]     = (_Float16)Arow[1][k0 + akb + j];
      a1[j + 8] = (_Float16)Arow[1][k0 + 16 + akb + j];
      a2[j]     = (_Float16)Arow[2][k0 + akb + j];
      a2[j + 8] = (_Float16)Arow[2][k0 + 16 + akb + j];
      a3[j]     = (_Float16)Arow[3][k0 + akb + j];
      a3[j + 8] = (_Float16)Arow[3][k0 + 16 + akb + j];
    }
    c0 = __builtin_amdgcn_wmma_f32_16x16x32_f16(false, a0, false, b, (short)0, c0, false, false);
    c1 = __builtin_amdgcn_wmma_f32_16x16x32_f16(false, a1, false, b, (short)0, c1, false, false);
    c2 = __builtin_amdgcn_wmma_f32_16x16x32_f16(false, a2, false, b, (short)0, c2, false, false);
    c3 = __builtin_amdgcn_wmma_f32_16x16x32_f16(false, a3, false, b, (short)0, c3, false, false);
  }

  const int col = n0 + l16;
  const float bv = bias[col];
  v8f acc[4] = {c0, c1, c2, c3};
#pragma unroll
  for (int i = 0; i < 4; ++i) {
    const int m0 = (tm0 + i) * 16;
#pragma unroll
    for (int j = 0; j < 8; ++j) {           // C/D: VGPR j -> row j + 8*half
      const int row = m0 + j + half * 8;
      if (row < M) {
        float val = acc[i][j] + bv;
        if (dogelu) val = 0.5f * val * (1.0f + erff(val * 0.70710678118f));
        if (R) val += R[(size_t)row * N + col];
        Y[(size_t)row * N + col] = val;
      }
    }
  }
}

// ------------------------- causal attention (wave per query) ---------------
// qkv layout: row (b*T+t), cols [0,3C): q | k | v.  hd in {32,64}.
__global__ void attn_kernel(const float* __restrict__ qkv,
                            float* __restrict__ y,
                            int T, int C, int hd) {
  const int qpos = blockIdx.x, h = blockIdx.y, b = blockIdx.z;
  const int lane = threadIdx.x;
  const bool two = (hd > 32);
  const float scale = rsqrtf((float)hd);

  const float* Q = qkv + (size_t)(b * T + qpos) * 3 * C + h * hd;
  float q0 = Q[lane];
  float q1 = two ? Q[lane + 32] : 0.f;

  float m = -1e30f, l = 0.f, acc0 = 0.f, acc1 = 0.f;
  for (int t = 0; t <= qpos; ++t) {
    const float* Kp = qkv + (size_t)(b * T + t) * 3 * C + C + h * hd;
    const float* Vp = Kp + C;
    float part = q0 * Kp[lane] + (two ? q1 * Kp[lane + 32] : 0.f);
    float s = wsum(part) * scale;
    float mn = fmaxf(m, s);
    float corr = __expf(m - mn);
    float e = __expf(s - mn);
    l = l * corr + e;
    acc0 = acc0 * corr + e * Vp[lane];
    if (two) acc1 = acc1 * corr + e * Vp[lane + 32];
    m = mn;
  }
  float* Yp = y + (size_t)(b * T + qpos) * C + h * hd;
  Yp[lane] = acc0 / l;
  if (two) Yp[lane + 32] = acc1 / l;
}

// ------------------------- global-token selection scan ---------------------
__global__ void select_kernel(const int* __restrict__ idx,
                              int* __restrict__ gts, int* __restrict__ gT) {
  if (threadIdx.x != 0) return;
  int b = blockIdx.x;
  int cnt = 0, total = 0, prevc = 0;
  for (int t = 0; t < TT; ++t) {
    int v = idx[b * TT + t];
    int c = (v < 48) || (v > 57 && v < 65) || (v > 90 && v < 97) ||
            (v > 122 && v < 128) || (v >= 192);
    int u = c && !(t > 0 && prevc);        // drop consecutive repeats
    if (u) {
      if (cnt < TGLB) gts[b * TGLB + cnt++] = t;
      ++total;
    }
    prevc = c;
  }
  for (int i = cnt; i < TGLB; ++i) gts[b * TGLB + i] = TT - 1;
  gT[b] = total < TGLB ? total : TGLB;
}

// ------------------------- gather / scatter --------------------------------
__global__ void gather_kernel(const float* __restrict__ x,
                              const int* __restrict__ gts,
                              const float* __restrict__ wpeg,
                              float* __restrict__ y) {
  int i = blockIdx.x * blockDim.x + threadIdx.x;   // B*TG*D
  if (i >= BB * TGLB * DGLB) return;
  int dd = i % DGLB;
  int bg = i / DGLB;
  int g = bg % TGLB;
  int b = bg / TGLB;
  int t = gts[b * TGLB + g];
  float val = (dd < DGLB - DLOC)
                  ? 0.f
                  : x[((size_t)b * TT + t) * DLOC + (dd - (DGLB - DLOC))];
  y[i] = val + wpeg[(size_t)g * DGLB + dd];
}

__global__ void scatter_kernel(float* __restrict__ x,
                               const float* __restrict__ y,
                               const int* __restrict__ gts,
                               const int* __restrict__ gT) {
  int i = blockIdx.x * blockDim.x + threadIdx.x;   // B*TG*DL
  if (i >= BB * TGLB * DLOC) return;
  int dd = i % DLOC;
  int bg = i / DLOC;
  int g = bg % TGLB;
  int b = bg / TGLB;
  if (g >= gT[b]) return;                          // valid positions only
  int t = gts[b * TGLB + g];
  x[((size_t)b * TT + t) * DLOC + dd] +=
      y[((size_t)b * TGLB + g) * DGLB + (DGLB - DLOC) + dd];
}

// ------------------------- logits GEMV (wave per vocab row) ----------------
__global__ void logits_kernel(const float* __restrict__ lastln,
                              const float* __restrict__ wte,
                              float* __restrict__ out) {
  int v = blockIdx.x, b = blockIdx.y, lane = threadIdx.x;
  const float* wr = wte + (size_t)v * DLOC;
  const float* xr = lastln + (size_t)b * DLOC;
  float s = 0.f;
  for (int d = lane; d < DLOC; d += 32) s += xr[d] * wr[d];
  s = wsum(s);
  if (lane == 0) out[(size_t)b * NVOCAB + v] = s;
}

// ---------------------------------------------------------------------------
// Host orchestration
// ---------------------------------------------------------------------------
struct BP {
  const float *ln1_w, *ln1_b, *attn_w, *attn_b, *proj_w, *proj_b;
  const float *ln2_w, *ln2_b, *fc_w, *fc_b, *fc2_w, *fc2_b;
};

static BP getBP(void* const* d_in, int base) {
  BP p;
  p.ln1_w = (const float*)d_in[base + 0];
  p.ln1_b = (const float*)d_in[base + 1];
  p.attn_w = (const float*)d_in[base + 2];
  p.attn_b = (const float*)d_in[base + 3];
  p.proj_w = (const float*)d_in[base + 4];
  p.proj_b = (const float*)d_in[base + 5];
  p.ln2_w = (const float*)d_in[base + 6];
  p.ln2_b = (const float*)d_in[base + 7];
  p.fc_w = (const float*)d_in[base + 8];
  p.fc_b = (const float*)d_in[base + 9];
  p.fc2_w = (const float*)d_in[base + 10];
  p.fc2_b = (const float*)d_in[base + 11];
  return p;
}

static void gemm_launch(const float* X, const float* W, const float* bias,
                        const float* R, float* Y, int M, int N, int K,
                        int gelu, hipStream_t s) {
  int tilesM = (M + 15) / 16;
  int tilesN = N / 16;
  dim3 grid((tilesN + 7) / 8, (tilesM + 3) / 4);
  gemm_wmma<<<grid, 256, 0, s>>>(X, W, bias, R, Y, M, N, K, gelu);
}

// One transformer block, faithful to the reference:
//   r = x; a = attn(ln1(x)); tmp = r + a; h = ln2(tmp);
//   m = fc2(gelu(fc(h)));   x_out = r + m        (r, not tmp!)
static void run_layer(const BP& p, int l, int d, int Trows, float* x,
                      float* xout, float* LNb, float* QKV, float* ATT,
                      float* TMP, float* LN2, float* FC, hipStream_t s) {
  const int Mr = BB * Trows;
  const int hd = d / NHEAD;
  const size_t dd = (size_t)d;

  layernorm_kernel<<<Mr, 32, 0, s>>>(x, p.ln1_w + l * dd, p.ln1_b + l * dd,
                                     LNb, d, d);
  gemm_launch(LNb, p.attn_w + l * dd * 3 * dd, p.attn_b + l * 3 * dd, nullptr,
              QKV, Mr, 3 * d, d, 0, s);
  dim3 ag(Trows, NHEAD, BB);
  attn_kernel<<<ag, 32, 0, s>>>(QKV, ATT, Trows, d, hd);
  gemm_launch(ATT, p.proj_w + l * dd * dd, p.proj_b + l * dd, x, TMP, Mr, d, d,
              0, s);
  layernorm_kernel<<<Mr, 32, 0, s>>>(TMP, p.ln2_w + l * dd, p.ln2_b + l * dd,
                                     LN2, d, d);
  gemm_launch(LN2, p.fc_w + l * dd * 4 * dd, p.fc_b + l * 4 * dd, nullptr, FC,
              Mr, 4 * d, d, 1, s);
  gemm_launch(FC, p.fc2_w + l * 4 * dd * dd, p.fc2_b + l * dd, x, xout, Mr, d,
              4 * d, 0, s);
}

extern "C" void kernel_launch(void* const* d_in, const int* in_sizes, int n_in,
                              void* d_out, int out_size, void* d_ws,
                              size_t ws_size, hipStream_t stream) {
  const int* idx = (const int*)d_in[0];
  const float* wte = (const float*)d_in[1];
  const float* wpe = (const float*)d_in[2];
  const float* wpe_global = (const float*)d_in[3];
  const float* lnf_w = (const float*)d_in[4];
  const float* lnf_b = (const float*)d_in[5];
  BP p_init = getBP(d_in, 6);
  BP p_glob = getBP(d_in, 18);
  BP p_fin = getBP(d_in, 30);

  // ---------------- workspace carve (floats) ----------------
  float* ws = (float*)d_ws;
  float* X0 = ws;                        // B*T*DL      = 524288
  float* X1 = X0 + 524288;               // 524288
  float* LNb = X1 + 524288;              // 524288 (>=170*1024 for global)
  float* QKV = LNb + 524288;             // 1572864 (B*T*3*DL; >=170*3072)
  float* ATT = QKV + 1572864;            // 524288
  float* TMP = ATT + 524288;             // 524288
  float* LN2 = TMP + 524288;             // 524288
  float* FC = LN2 + 524288;              // 2097152 (B*T*4*DL; >=170*4096)
  float* Y0 = FC + 2097152;              // 262144 (>=B*TG*D=174080)
  float* Y1 = Y0 + 262144;               // 262144
  float* LASTLN = Y1 + 262144;           // 1024 (B*DL)
  int* gts = (int*)(LASTLN + 1024);      // B*TG ints
  int* gT = gts + BB * TGLB;             // B ints

  // ---------------- embedding ----------------
  {
    int total = BB * TT * DLOC;
    embed_kernel<<<(total + 255) / 256, 256, 0, stream>>>(idx, wte, wpe, X0);
  }

  // ---------------- local blocks (init) ----------------
  float* xc = X0;
  float* xa = X1;
  for (int l = 0; l < NLH; ++l) {
    run_layer(p_init, l, DLOC, TT, xc, xa, LNb, QKV, ATT, TMP, LN2, FC, stream);
    float* t = xc; xc = xa; xa = t;
  }

  // ---------------- select global tokens, gather ----------------
  select_kernel<<<BB, 32, 0, stream>>>(idx, gts, gT);
  {
    int total = BB * TGLB * DGLB;
    gather_kernel<<<(total + 255) / 256, 256, 0, stream>>>(xc, gts, wpe_global,
                                                           Y0);
  }

  // ---------------- global blocks ----------------
  float* yc = Y0;
  float* ya = Y1;
  for (int l = 0; l < NLG; ++l) {
    run_layer(p_glob, l, DGLB, TGLB, yc, ya, LNb, QKV, ATT, TMP, LN2, FC,
              stream);
    float* t = yc; yc = ya; ya = t;
  }

  // ---------------- scatter-add back into local stream ----------------
  {
    int total = BB * TGLB * DLOC;
    scatter_kernel<<<(total + 255) / 256, 256, 0, stream>>>(xc, yc, gts, gT);
  }

  // ---------------- local blocks (final) ----------------
  for (int l = 0; l < NLH; ++l) {
    run_layer(p_fin, l, DLOC, TT, xc, xa, LNb, QKV, ATT, TMP, LN2, FC, stream);
    float* t = xc; xc = xa; xa = t;
  }

  // ---------------- final LN (last token only) + logits ----------------
  layernorm_kernel<<<BB, 32, 0, stream>>>(xc + (size_t)(TT - 1) * DLOC, lnf_w,
                                          lnf_b, LASTLN, DLOC,
                                          (long)TT * DLOC);
  dim3 lg(NVOCAB, BB);
  logits_kernel<<<lg, 32, 0, stream>>>(LASTLN, wte, (float*)d_out);
}
